// TumorGraphGNN_2302102471068
// MI455X (gfx1250) — matrix operand order repaired and checked by reading, not compiled, not run on hardware
//
#include <hip/hip_runtime.h>
#include <hip/hip_bf16.h>

// ---------------- problem constants (match reference) ----------------
#define NN     50000
#define NE     1600000
#define NG     512
#define DIN    128
#define HDIM   256     // H1 == H2 == 256
#define DEMB   128
#define BN_EPS 1e-5f

typedef __attribute__((ext_vector_type(16))) __bf16 v16bf;
typedef __attribute__((ext_vector_type(8)))  float  v8f;

union ABFrag { unsigned u[8]; v16bf v; };

__device__ __forceinline__ unsigned short f2bf(float f) {
    union { float f; unsigned u; } x; x.f = f;
    unsigned r = x.u + 0x7FFFu + ((x.u >> 16) & 1u);   // round-to-nearest-even
    return (unsigned short)(r >> 16);
}

// ---------------- small utility kernels ----------------
__global__ __launch_bounds__(256) void k_init_f32(float* p, int n, float v) {
    int i = blockIdx.x * 256 + threadIdx.x;
    if (i < n) p[i] = v;
}

__global__ __launch_bounds__(256) void k_edge_deg(const int* __restrict__ dst, float* __restrict__ deg, int E) {
    int i = blockIdx.x * 256 + threadIdx.x;
    if (i < E) unsafeAtomicAdd(&deg[dst[i]], 1.0f);
}

__global__ __launch_bounds__(256) void k_dinv(float* __restrict__ deg, int n) {
    int i = blockIdx.x * 256 + threadIdx.x;
    if (i < n) deg[i] = rsqrtf(deg[i]);   // deg >= 1 due to self loop
}

__global__ __launch_bounds__(256) void k_f32_to_bf16(const float* __restrict__ in, unsigned short* __restrict__ out, int n) {
    int i = blockIdx.x * 256 + threadIdx.x;
    if (i < n) out[i] = f2bf(in[i]);
}

// Pack W [K x N] row-major fp32 into per-lane WMMA B fragments:
// element index ((nt*KT + kt)*32 + lane)*16 .. +15  (16 bf16 = one lane's v16bf)
// B k order used here, matched by A (bf16 32x16): lanes 0-15 -> K=0..15 (VGPR j: K=2j,2j+1), lanes 16-31 -> K=16..31
__global__ __launch_bounds__(256) void k_pack_b(const float* __restrict__ W, unsigned short* __restrict__ Bp,
                                                int K, int N) {
    int KT = K >> 5, NT = N >> 4;
    int id = blockIdx.x * 256 + threadIdx.x;
    if (id >= NT * KT * 32) return;
    int lane = id & 31;
    int kt   = (id >> 5) % KT;
    int nt   = (id >> 5) / KT;
    int n     = nt * 16 + (lane & 15);
    int khalf = lane >> 4;
    unsigned short* o = Bp + (size_t)id * 16;
#pragma unroll
    for (int j = 0; j < 8; ++j) {
        int k = kt * 32 + khalf * 16 + 2 * j;
        o[2 * j]     = f2bf(W[(size_t)k * N + n]);
        o[2 * j + 1] = f2bf(W[(size_t)(k + 1) * N + n]);
    }
}

// ---------------- WMMA GEMM: C[M x NT*16] = A[M x K](bf16) * B(packed) (+bias) ----------------
// grid.x = M/16 (exact), block = 256 (8 waves); wave handles ntiles wave, wave+8, ...
__global__ __launch_bounds__(256) void k_gemm_wmma(const unsigned short* __restrict__ A,
                                                   const unsigned short* __restrict__ Bp,
                                                   float* __restrict__ C,
                                                   int K, int NT, const float* __restrict__ bias) {
    const int KT    = K >> 5;
    const int mtile = blockIdx.x;
    const int wave  = threadIdx.x >> 5;
    const int lane  = threadIdx.x & 31;
    const int lrow  = lane & 15;
    const int khalf = lane >> 4;
    const int ldc   = NT << 4;
    const unsigned short* arow = A + (size_t)(mtile * 16 + lrow) * K;

    for (int nt = wave; nt < NT; nt += 8) {
        v8f c = {};
        const unsigned short* bptr = Bp + ((size_t)nt * KT * 32 + lane) * 16;
        for (int kt = 0; kt < KT; ++kt) {
            ABFrag a, b;
#pragma unroll
            for (int j = 0; j < 8; ++j) {
                // A k order matches packed B: lanes0-15 -> K=2j,2j+1; lanes16-31 +16
                int k0 = kt * 32 + khalf * 16 + 2 * j;
                a.u[j] = *(const unsigned*)(arow + k0);
            }
            const unsigned* bs = (const unsigned*)(bptr + (size_t)kt * 32 * 16);
#pragma unroll
            for (int j = 0; j < 8; ++j) b.u[j] = bs[j];
            c = __builtin_amdgcn_wmma_f32_16x16x32_bf16(false, a.v, false, b.v, (short)0, c, false, false);
        }
        int col = nt * 16 + lrow;
        float bv = bias ? bias[col] : 0.0f;
#pragma unroll
        for (int r = 0; r < 8; ++r) {
            int m = mtile * 16 + r + 8 * khalf;   // D layout: VGPR r -> M = r + 8*(lane>>4)
            C[(size_t)m * ldc + col] = c[r] + bv;
        }
    }
}

// ---------------- sparse aggregation ----------------
// 64 threads per edge, 4 features each (float4 gather, 4x atomic_add_f32 scatter into L2)
__global__ __launch_bounds__(256) void k_edge_scatter(const int* __restrict__ src, const int* __restrict__ dst,
                                                      const float* __restrict__ dinv, const float* __restrict__ X,
                                                      float* __restrict__ H, int E) {
    long long g = (long long)blockIdx.x * 256 + threadIdx.x;
    int e = (int)(g >> 6);
    if (e >= E) return;
    int f = ((int)g & 63) << 2;
    int s = src[e], d = dst[e];
    float w = dinv[s] * dinv[d];
    float4 x = *(const float4*)(X + (size_t)s * HDIM + f);
    float* h = H + (size_t)d * HDIM + f;
    unsafeAtomicAdd(h + 0, x.x * w);
    unsafeAtomicAdd(h + 1, x.y * w);
    unsafeAtomicAdd(h + 2, x.z * w);
    unsafeAtomicAdd(h + 3, x.w * w);
}

// self-loop term + conv bias: H[i,c] += dinv[i]^2 * X[i,c] + b[c]
__global__ __launch_bounds__(256) void k_self_bias(const float* __restrict__ dinv, const float* __restrict__ X,
                                                   const float* __restrict__ b, float* __restrict__ H, int total) {
    int g = blockIdx.x * 256 + threadIdx.x;
    if (g >= total) return;
    int col = g & (HDIM - 1);
    int node = g >> 8;
    float di = dinv[node];
    H[g] += di * di * X[g] + b[col];
}

// ---------------- batch norm ----------------
__global__ __launch_bounds__(256) void k_bn_stats(const float* __restrict__ H, float* __restrict__ s,
                                                  float* __restrict__ s2, int nrows) {
    int col = threadIdx.x;            // 256 columns
    int r0 = blockIdx.x * 256;
    float a = 0.f, q = 0.f;
    for (int i = 0; i < 256; ++i) {
        int r = r0 + i;
        if (r >= nrows) break;
        float v = H[(size_t)r * HDIM + col];
        a += v; q += v * v;
    }
    unsafeAtomicAdd(&s[col], a);
    unsafeAtomicAdd(&s2[col], q);
}

__device__ __forceinline__ float bn_val(float v, int col, const float* s, const float* s2,
                                        const float* gm, const float* bt, float invN) {
    float mean = s[col] * invN;
    float var  = s2[col] * invN - mean * mean;
    float y = (v - mean) * rsqrtf(var + BN_EPS) * gm[col] + bt[col];
    return y > 0.f ? y : 0.f;   // ReLU
}

__global__ __launch_bounds__(256) void k_bn_relu_bf16(const float* __restrict__ H, const float* __restrict__ s,
                                                      const float* __restrict__ s2, const float* __restrict__ gm,
                                                      const float* __restrict__ bt, unsigned short* __restrict__ out,
                                                      int total, float invN) {
    int g = blockIdx.x * 256 + threadIdx.x;
    if (g >= total) return;
    out[g] = f2bf(bn_val(H[g], g & (HDIM - 1), s, s2, gm, bt, invN));
}

// BN+ReLU fused with global mean-pool accumulation
__global__ __launch_bounds__(256) void k_bn_relu_pool(const float* __restrict__ H, const float* __restrict__ s,
                                                      const float* __restrict__ s2, const float* __restrict__ gm,
                                                      const float* __restrict__ bt, const int* __restrict__ bid,
                                                      float* __restrict__ Gsum, int total, float invN) {
    int g = blockIdx.x * 256 + threadIdx.x;
    if (g >= total) return;
    int col = g & (HDIM - 1);
    int node = g >> 8;
    float y = bn_val(H[g], col, s, s2, gm, bt, invN);
    unsafeAtomicAdd(&Gsum[(size_t)bid[node] * HDIM + col], y);
}

__global__ __launch_bounds__(256) void k_counts(const int* __restrict__ bid, float* __restrict__ cnt, int n) {
    int i = blockIdx.x * 256 + threadIdx.x;
    if (i < n) unsafeAtomicAdd(&cnt[bid[i]], 1.0f);
}

__global__ __launch_bounds__(256) void k_pool_finalize(const float* __restrict__ Gsum, const float* __restrict__ cnt,
                                                       unsigned short* __restrict__ Gbf, int total) {
    int i = blockIdx.x * 256 + threadIdx.x;
    if (i >= total) return;
    float c = cnt[i >> 8];
    Gbf[i] = f2bf(Gsum[i] / (c > 1.0f ? c : 1.0f));
}

// ---------------- host-side launch ----------------
static inline int cdiv(long long a, long long b) { return (int)((a + b - 1) / b); }

extern "C" void kernel_launch(void* const* d_in, const int* in_sizes, int n_in,
                              void* d_out, int out_size, void* d_ws, size_t ws_size,
                              hipStream_t stream) {
    (void)in_sizes; (void)n_in; (void)out_size; (void)ws_size;
    const float* V   = (const float*)d_in[0];
    const int*   ei  = (const int*)d_in[1];
    const int*   bid = (const int*)d_in[2];
    const float* W1  = (const float*)d_in[3];
    const float* b1  = (const float*)d_in[4];
    const float* g1  = (const float*)d_in[5];
    const float* be1 = (const float*)d_in[6];
    const float* W2  = (const float*)d_in[7];
    const float* b2  = (const float*)d_in[8];
    const float* g2  = (const float*)d_in[9];
    const float* be2 = (const float*)d_in[10];
    const float* Wfc = (const float*)d_in[11];
    const float* bfc = (const float*)d_in[12];
    float* out = (float*)d_out;

    const int* src = ei;        // edge_index[0]
    const int* dst = ei + NE;   // edge_index[1]

    // workspace carve-out (~130 MB)
    char* base = (char*)d_ws;
    size_t off = 0;
    auto carve = [&](size_t bytes) -> char* {
        char* p = base + off;
        off = (off + bytes + 255) & ~(size_t)255;
        return p;
    };
    float*          deg  = (float*)carve((size_t)NN * 4);                 // -> dinv in place
    float*          XW   = (float*)carve((size_t)NN * HDIM * 4);
    float*          Hb   = (float*)carve((size_t)NN * HDIM * 4);
    unsigned short* Abf  = (unsigned short*)carve((size_t)NN * HDIM * 2);
    unsigned short* Bp1  = (unsigned short*)carve((size_t)(DIN / 32) * (HDIM / 16) * 512 * 2);
    unsigned short* Bp2  = (unsigned short*)carve((size_t)(HDIM / 32) * (HDIM / 16) * 512 * 2);
    unsigned short* Bpf  = (unsigned short*)carve((size_t)(HDIM / 32) * (DEMB / 16) * 512 * 2);
    float*          cs   = (float*)carve(HDIM * 4);
    float*          cs2  = (float*)carve(HDIM * 4);
    float*          Gsum = (float*)carve((size_t)NG * HDIM * 4);
    float*          cnt  = (float*)carve((size_t)NG * 4);
    unsigned short* Gbf  = (unsigned short*)carve((size_t)NG * HDIM * 2);

    const float invN = 1.0f / (float)NN;
    const int NEL = NN * HDIM;   // 12.8M, fits int

    // degrees (self-loop => start at 1), then dinv
    k_init_f32<<<cdiv(NN, 256), 256, 0, stream>>>(deg, NN, 1.0f);
    k_edge_deg<<<cdiv(NE, 256), 256, 0, stream>>>(dst, deg, NE);
    k_dinv<<<cdiv(NN, 256), 256, 0, stream>>>(deg, NN);

    // pack weights into WMMA fragments
    k_pack_b<<<cdiv((DIN / 32) * (HDIM / 16) * 32, 256), 256, 0, stream>>>(W1, Bp1, DIN, HDIM);
    k_pack_b<<<cdiv((HDIM / 32) * (HDIM / 16) * 32, 256), 256, 0, stream>>>(W2, Bp2, HDIM, HDIM);
    k_pack_b<<<cdiv((HDIM / 32) * (DEMB / 16) * 32, 256), 256, 0, stream>>>(Wfc, Bpf, HDIM, DEMB);

    // ---- layer 1 ----
    k_f32_to_bf16<<<cdiv(NN * DIN, 256), 256, 0, stream>>>(V, Abf, NN * DIN);
    k_gemm_wmma<<<NN / 16, 256, 0, stream>>>(Abf, Bp1, XW, DIN, HDIM / 16, nullptr);
    k_init_f32<<<cdiv(NEL, 256), 256, 0, stream>>>(Hb, NEL, 0.0f);
    k_edge_scatter<<<cdiv((long long)NE * 64, 256), 256, 0, stream>>>(src, dst, deg, XW, Hb, NE);
    k_self_bias<<<cdiv(NEL, 256), 256, 0, stream>>>(deg, XW, b1, Hb, NEL);
    k_init_f32<<<cdiv(HDIM, 256), 256, 0, stream>>>(cs, HDIM, 0.0f);
    k_init_f32<<<cdiv(HDIM, 256), 256, 0, stream>>>(cs2, HDIM, 0.0f);
    k_bn_stats<<<cdiv(NN, 256), 256, 0, stream>>>(Hb, cs, cs2, NN);
    k_bn_relu_bf16<<<cdiv(NEL, 256), 256, 0, stream>>>(Hb, cs, cs2, g1, be1, Abf, NEL, invN);

    // ---- layer 2 ----
    k_gemm_wmma<<<NN / 16, 256, 0, stream>>>(Abf, Bp2, XW, HDIM, HDIM / 16, nullptr);
    k_init_f32<<<cdiv(NEL, 256), 256, 0, stream>>>(Hb, NEL, 0.0f);
    k_edge_scatter<<<cdiv((long long)NE * 64, 256), 256, 0, stream>>>(src, dst, deg, XW, Hb, NE);
    k_self_bias<<<cdiv(NEL, 256), 256, 0, stream>>>(deg, XW, b2, Hb, NEL);
    k_init_f32<<<cdiv(HDIM, 256), 256, 0, stream>>>(cs, HDIM, 0.0f);
    k_init_f32<<<cdiv(HDIM, 256), 256, 0, stream>>>(cs2, HDIM, 0.0f);
    k_bn_stats<<<cdiv(NN, 256), 256, 0, stream>>>(Hb, cs, cs2, NN);

    // ---- pooling + FC ----
    k_init_f32<<<cdiv(NG * HDIM, 256), 256, 0, stream>>>(Gsum, NG * HDIM, 0.0f);
    k_init_f32<<<cdiv(NG, 256), 256, 0, stream>>>(cnt, NG, 0.0f);
    k_counts<<<cdiv(NN, 256), 256, 0, stream>>>(bid, cnt, NN);
    k_bn_relu_pool<<<cdiv(NEL, 256), 256, 0, stream>>>(Hb, cs, cs2, g2, be2, bid, Gsum, NEL, invN);
    k_pool_finalize<<<cdiv(NG * HDIM, 256), 256, 0, stream>>>(Gsum, cnt, Gbf, NG * HDIM);
    k_gemm_wmma<<<NG / 16, 256, 0, stream>>>(Gbf, Bpf, out, HDIM, DEMB / 16, bfc);
}
